// CondGINConv_39247411151301
// MI455X (gfx1250) — matrix operand order repaired and checked by reading, not compiled, not run on hardware
//
#include <hip/hip_runtime.h>
#include <hip/hip_bf16.h>

// ---------------------------------------------------------------------------
// CondGINConv for MI455X (gfx1250, wave32, WMMA)
//   phase 0: split W1/W2 once into bf16 hi/lo, transposed [n][k] (L2-resident)
//   phase 1: key = condition @ Wk
//   phase 2: h = x ; h[row] += x[col] * sigmoid(leaky(x[col].ki + x[row].kj))
//   phase 3: out = relu(h@W1+b1)@W2+b2   (split-bf16 WMMA, pipelined)
// ---------------------------------------------------------------------------

#define N_NODES 50000
#define N_EDGES 400000
#define DD      512
#define HH      1024

typedef __attribute__((ext_vector_type(8)))  float  v8f;
typedef __attribute__((ext_vector_type(4)))  __bf16 v4bf;
typedef __attribute__((ext_vector_type(8)))  __bf16 v8bf;
typedef __attribute__((ext_vector_type(16))) __bf16 v16bf;

#define BM 128
#define BN 128
#define BK 32
#define LDSTR 40   // padded LDS row stride in bf16 elems (80B: 16B-aligned, bank-friendly)

// ---------------------------------------------------------------------------
// One-time weight split: W (K x N, f32) -> hi/lo bf16 stored transposed [n][k]
// ---------------------------------------------------------------------------
__global__ void split_w(const float* __restrict__ W,
                        __bf16* __restrict__ hi, __bf16* __restrict__ lo,
                        int K, int N) {
  int idx = blockIdx.x * blockDim.x + threadIdx.x;   // idx over [N][K]
  if (idx >= K * N) return;
  int n = idx / K;
  int k = idx % K;                                    // consecutive idx -> coalesced writes
  float f = W[(size_t)k * N + n];
  __bf16 hb = (__bf16)f;
  hi[idx] = hb;
  lo[idx] = (__bf16)(f - (float)hb);
}

// ---------------------------------------------------------------------------
// key = condition(1,256) @ Wk(256, 1024)
// ---------------------------------------------------------------------------
__global__ void key_kernel(const float* __restrict__ cond,
                           const float* __restrict__ Wk,
                           float* __restrict__ key) {
  int c = blockIdx.x * blockDim.x + threadIdx.x;
  if (c >= HH) return;
  float acc = 0.f;
  #pragma unroll 4
  for (int r = 0; r < 256; ++r) acc += cond[r] * Wk[r * HH + c];
  key[c] = acc;
}

// ---------------------------------------------------------------------------
// One wave32 per edge: dot products + sigmoid(leaky) + atomic scatter-add.
// x (102MB) is L2-resident on MI455X (192MB L2) -> gathers hit L2.
// ---------------------------------------------------------------------------
__launch_bounds__(256)
__global__ void edge_kernel(const float* __restrict__ x,
                            const int* __restrict__ ei,
                            const float* __restrict__ key,
                            float* __restrict__ h) {
  int gt   = blockIdx.x * blockDim.x + threadIdx.x;
  int e    = gt >> 5;
  int lane = threadIdx.x & 31;
  if (e >= N_EDGES) return;
  int r = ei[e];             // edge_index[0]
  int c = ei[N_EDGES + e];   // edge_index[1]

  const float4* xc = (const float4*)(x + (size_t)c * DD) + lane * 4;
  const float4* xr = (const float4*)(x + (size_t)r * DD) + lane * 4;
  const float4* ki = (const float4*)(key)      + lane * 4;
  const float4* kj = (const float4*)(key + DD) + lane * 4;

  float4 xcv[4];
  float acc = 0.f;
  #pragma unroll
  for (int i = 0; i < 4; ++i) {
    float4 a = xc[i], b = ki[i], p = xr[i], q = kj[i];
    xcv[i] = a;
    acc += a.x * b.x + a.y * b.y + a.z * b.z + a.w * b.w;
    acc += p.x * q.x + p.y * q.y + p.z * q.z + p.w * q.w;
  }
  #pragma unroll
  for (int off = 16; off > 0; off >>= 1)
    acc += __shfl_xor(acc, off, 32);

  float al = (acc >= 0.f) ? acc : 0.2f * acc;       // leaky relu, slope 0.2
  al = 1.0f / (1.0f + __expf(-al));                 // sigmoid

  float* dst = h + (size_t)r * DD + lane * 16;
  #pragma unroll
  for (int i = 0; i < 4; ++i) {
    atomicAdd(dst + 4 * i + 0, xcv[i].x * al);
    atomicAdd(dst + 4 * i + 1, xcv[i].y * al);
    atomicAdd(dst + 4 * i + 2, xcv[i].z * al);
    atomicAdd(dst + 4 * i + 3, xcv[i].w * al);
  }
}

// ---------------------------------------------------------------------------
// C[M,N] = act(A[M,K] @ B[K,N] + bias); A fp32 (split on the fly, via LDS),
// B pre-split bf16 hi/lo transposed [n][k] (fragments loaded straight from
// global / L2). 3-term split: hiA*hiB + hiA*loB + loA*hiB, f32 accumulate.
// 256 threads = 8 waves (2x4); block tile 128x128x32; wave tile 64x32.
// Pipelined: next A tile + current B fragments are in flight across barrier.
// ---------------------------------------------------------------------------
__launch_bounds__(256)
__global__ void gemm_bias_act(const float* __restrict__ A,
                              const __bf16* __restrict__ Bthi,
                              const __bf16* __restrict__ Btlo,
                              const float* __restrict__ bias,
                              float* __restrict__ C,
                              int M, int N, int K, int do_relu) {
  __shared__ __bf16 sAhi[BM * LDSTR];
  __shared__ __bf16 sAlo[BM * LDSTR];

  const int t    = threadIdx.x;
  const int lane = t & 31;
  const int w    = t >> 5;            // wave 0..7
  const int wm   = (w >> 2) * 64;     // wave row base within block tile
  const int wn   = (w & 3) * 32;      // wave col base within block tile
  const int half = lane >> 4;         // 0 = lanes 0-15, 1 = lanes 16-31
  const int l15  = lane & 15;

  const int m0 = blockIdx.x * BM;
  const int n0 = blockIdx.y * BN;

  // A staging coords: each thread owns half a row (16 floats)
  const int ar = t >> 1;
  const int ac = (t & 1) * 16;
  const int gm = m0 + ar;
  const bool arow_ok = (gm < M);
  const float* asrc = A + (size_t)gm * K + ac;

  v8f acc[4][2] = {};

  // per-lane B fragment base pointers (Bt layout [n][k], row stride K)
  const __bf16* bhp[2];
  const __bf16* blp[2];
  #pragma unroll
  for (int nt = 0; nt < 2; ++nt) {
    int gn = n0 + wn + nt * 16 + l15;
    bhp[nt] = Bthi + (size_t)gn * K + half * 8;
    blp[nt] = Btlo + (size_t)gn * K + half * 8;
  }

  union Fr { v16bf v; v8bf h[2]; };

  // prologue: A tile for k0=0 into registers
  float4 areg[4];
  #pragma unroll
  for (int i = 0; i < 4; ++i)
    areg[i] = arow_ok ? *(const float4*)(asrc + 4 * i)
                      : make_float4(0.f, 0.f, 0.f, 0.f);

  for (int k0 = 0; k0 < K; k0 += BK) {
    // ---- convert current A regs -> bf16 hi/lo ----
    v4bf hi[4], lo[4];
    #pragma unroll
    for (int i = 0; i < 4; ++i) {
      float f[4] = {areg[i].x, areg[i].y, areg[i].z, areg[i].w};
      #pragma unroll
      for (int j = 0; j < 4; ++j) {
        __bf16 hb = (__bf16)f[j];
        hi[i][j] = hb;
        lo[i][j] = (__bf16)(f[j] - (float)hb);
      }
    }
    // ---- issue next A tile loads (in flight across compute) ----
    if (k0 + BK < K) {
      #pragma unroll
      for (int i = 0; i < 4; ++i)
        areg[i] = arow_ok ? *(const float4*)(asrc + (k0 + BK) + 4 * i)
                          : make_float4(0.f, 0.f, 0.f, 0.f);
    }
    // ---- store converted A tile to LDS ----
    #pragma unroll
    for (int i = 0; i < 4; ++i) {
      *(v4bf*)&sAhi[ar * LDSTR + ac + 4 * i] = hi[i];
      *(v4bf*)&sAlo[ar * LDSTR + ac + 4 * i] = lo[i];
    }
    // ---- issue B fragment loads for this k-step (global, L2-resident) ----
    Fr bhi[2], blo[2];
    #pragma unroll
    for (int nt = 0; nt < 2; ++nt) {
      bhi[nt].h[0] = *(const v8bf*)(bhp[nt] + k0);
      bhi[nt].h[1] = *(const v8bf*)(bhp[nt] + k0 + 16);
      blo[nt].h[0] = *(const v8bf*)(blp[nt] + k0);
      blo[nt].h[1] = *(const v8bf*)(blp[nt] + k0 + 16);
    }
    __syncthreads();

    // ---- A fragments from LDS per ISA 16-bit striping ----
    Fr ahi[4], alo[4];
    const int ks0 = half ? 8  : 0;   // lanes 0-15: K 0-7/16-23; lanes 16-31: K 8-15/24-31
    const int ks1 = half ? 24 : 16;
    #pragma unroll
    for (int mt = 0; mt < 4; ++mt) {
      int m = wm + mt * 16 + l15;
      ahi[mt].h[0] = *(const v8bf*)&sAhi[m * LDSTR + ks0];
      ahi[mt].h[1] = *(const v8bf*)&sAhi[m * LDSTR + ks1];
      alo[mt].h[0] = *(const v8bf*)&sAlo[m * LDSTR + ks0];
      alo[mt].h[1] = *(const v8bf*)&sAlo[m * LDSTR + ks1];
    }

    // ---- 24 WMMAs per K-step ----
    #pragma unroll
    for (int mt = 0; mt < 4; ++mt)
      #pragma unroll
      for (int nt = 0; nt < 2; ++nt) {
        acc[mt][nt] = __builtin_amdgcn_wmma_f32_16x16x32_bf16(
            false, ahi[mt].v, false, bhi[nt].v, (short)0, acc[mt][nt], false, false);
        acc[mt][nt] = __builtin_amdgcn_wmma_f32_16x16x32_bf16(
            false, ahi[mt].v, false, blo[nt].v, (short)0, acc[mt][nt], false, false);
        acc[mt][nt] = __builtin_amdgcn_wmma_f32_16x16x32_bf16(
            false, alo[mt].v, false, bhi[nt].v, (short)0, acc[mt][nt], false, false);
      }
    __syncthreads();
  }

  // ---- epilogue: bias (+ relu); D layout: vgpr r -> M = half*8 + r, N = l15 ----
  #pragma unroll
  for (int nt = 0; nt < 2; ++nt) {
    int gn = n0 + wn + nt * 16 + l15;
    float bv = bias[gn];
    #pragma unroll
    for (int mt = 0; mt < 4; ++mt) {
      #pragma unroll
      for (int r = 0; r < 8; ++r) {
        int gmm = m0 + wm + mt * 16 + half * 8 + r;
        if (gmm < M) {
          float v = acc[mt][nt][r] + bv;
          if (do_relu) v = v > 0.f ? v : 0.f;
          C[(size_t)gmm * N + gn] = v;
        }
      }
    }
  }
}

// ---------------------------------------------------------------------------
extern "C" void kernel_launch(void* const* d_in, const int* in_sizes, int n_in,
                              void* d_out, int out_size, void* d_ws, size_t ws_size,
                              hipStream_t stream) {
  const float* x    = (const float*)d_in[0];
  const int*   ei   = (const int*)  d_in[1];
  const float* cond = (const float*)d_in[2];
  const float* Wk   = (const float*)d_in[3];
  const float* W1   = (const float*)d_in[4];
  const float* b1   = (const float*)d_in[5];
  const float* W2   = (const float*)d_in[6];
  const float* b2   = (const float*)d_in[7];
  float* out = (float*)d_out;

  float* ws   = (float*)d_ws;
  float* key  = ws;                               // 1024 f32
  float* h    = ws + 1024;                        // 50000*512 f32
  float* hid  = h + (size_t)N_NODES * DD;         // 50000*1024 f32
  __bf16* w1hi = (__bf16*)(hid + (size_t)N_NODES * HH);
  __bf16* w1lo = w1hi + (size_t)DD * HH;
  __bf16* w2hi = w1lo + (size_t)DD * HH;
  __bf16* w2lo = w2hi + (size_t)HH * DD;

  // phase 0: one-time weight split (bf16 hi/lo, transposed [n][k])
  int wElems = DD * HH;
  split_w<<<(wElems + 255) / 256, 256, 0, stream>>>(W1, w1hi, w1lo, DD, HH);
  split_w<<<(wElems + 255) / 256, 256, 0, stream>>>(W2, w2hi, w2lo, HH, DD);

  // phase 1: key = condition @ Wk
  key_kernel<<<(HH + 255) / 256, 256, 0, stream>>>(cond, Wk, key);

  // phase 2: h = x (EPS==0), then edge scatter-add
  hipMemcpyAsync(h, x, (size_t)N_NODES * DD * sizeof(float),
                 hipMemcpyDeviceToDevice, stream);
  edge_kernel<<<(N_EDGES * 32 + 255) / 256, 256, 0, stream>>>(x, ei, key, h);

  // phase 3: MLP via split-bf16 WMMA GEMMs
  dim3 g1((N_NODES + BM - 1) / BM, HH / BN);
  gemm_bias_act<<<g1, 256, 0, stream>>>(h, w1hi, w1lo, b1, hid, N_NODES, HH, DD, 1);
  dim3 g2((N_NODES + BM - 1) / BM, DD / BN);
  gemm_bias_act<<<g2, 256, 0, stream>>>(hid, w2hi, w2lo, b2, out, N_NODES, DD, HH, 0);
}
